// MoEAttention_88321707475475
// MI455X (gfx1250) — matrix operand order repaired and checked by reading, not compiled, not run on hardware
//
#include <hip/hip_runtime.h>
#include <hip/hip_bf16.h>

// ---------------------------------------------------------------------------
// MoEAttention pipeline for MI455X (gfx1250, wave32, WMMA f32=f16xf16+f32)
// All GEMM-shaped math routed through v_wmma_f32_16x16x32_f16; all LDS
// fragment traffic through ds_load_b128/ds_store_b128-friendly layouts.
// ---------------------------------------------------------------------------

typedef __attribute__((ext_vector_type(16))) _Float16 v16h;
typedef __attribute__((ext_vector_type(8)))  _Float16 v8h;
typedef __attribute__((ext_vector_type(8)))  float    v8f;

#define NSITES   1024
#define NALLELES 2048
#define CIN      16
#define CC       32
#define LL       150
#define LCP      75      // compressed length
#define RPA      16      // reads per allele
#define KSTRIDE  104     // im2col K stride: 96 used, padded (208B, 16B-aligned)

__device__ __forceinline__ v8f wmma_f16(v16h a, v16h b, v8f c) {
  // D(16x16 f32) = A(16x32 f16) * B(32x16 f16) + C
  return __builtin_amdgcn_wmma_f32_16x16x32_f16(false, a, false, b, (short)0, c,
                                                false, false);
}

// A-matrix K index for vector element h given lane half (contiguous: K = h + 16*hf)
__device__ __forceinline__ int akidx(int h, int hf) {
  return h + (hf ? 16 : 0);
}

// two 16B LDS loads -> one 16-half fragment
__device__ __forceinline__ v16h ld2x8h(const _Float16* p0, const _Float16* p1) {
  v8h a = *(const v8h*)p0;
  v8h b = *(const v8h*)p1;
  return __builtin_shufflevector(a, b, 0, 1, 2, 3, 4, 5, 6, 7,
                                       8, 9, 10, 11, 12, 13, 14, 15);
}
__device__ __forceinline__ v16h ld16h(const _Float16* p) { return ld2x8h(p, p + 8); }

// pack 8 f32 accumulator elements to 8 halves (for one ds_store_b128)
__device__ __forceinline__ v8h cvt8h(v8f c) {
  v8h p;
#pragma unroll
  for (int j = 0; j < 8; ++j) p[j] = (_Float16)c[j];
  return p;
}

// ---------------------------------------------------------------------------
// Kernel 1: red[a,c,l] = sum_{r<16} relu(conv1d(x[a*16+r], w, k=5, pad=2))
// One block per allele, 8 waves. im2col GEMM: M=32(out ch), K=80->96, N=160.
// Register-pressure aware split: each wave owns ONE mt (3 A-fragments, 24
// VGPRs) and <=3 N-tiles (<=3 live accumulators) -> no scratch spills.
// LDS holds the im2col panel bm[col][K]: B fragments = two ds_load_b128.
// ---------------------------------------------------------------------------
__global__ __launch_bounds__(256)
void conv_reduce_kernel(const float* __restrict__ x,   // [Nr,16,150]
                        const float* __restrict__ w,   // [32,16,5]
                        float* __restrict__ red)       // [Na,32,150]
{
  const int a    = blockIdx.x;
  const int tid  = threadIdx.x;
  const int wid  = tid >> 5;          // 0..7
  const int lane = tid & 31;
  const int hf   = lane >> 4;
  const int l15  = lane & 15;
  const int mt   = wid >> 2;          // waves 0-3 -> mt0, waves 4-7 -> mt1
  const int bnt  = wid & 3;           // base N-tile; tiles: bnt, bnt+4, bnt+8

  __shared__ __align__(16) _Float16 bm[160][KSTRIDE];   // bm[l][i*5+t]

  // Zero entire panel once (pad rows/cols stay zero for all 16 reads).
  {
    uint4* p4 = (uint4*)&bm[0][0];
    const int n4 = 160 * KSTRIDE * (int)sizeof(_Float16) / 16;   // 2080
    uint4 z; z.x = 0u; z.y = 0u; z.z = 0u; z.w = 0u;
    for (int idx = tid; idx < n4; idx += 256) p4[idx] = z;
  }

  // Weight A-fragments for this wave's mt only: 3 K-tiles (24 VGPRs)
  v16h A[3];
#pragma unroll
  for (int kt = 0; kt < 3; ++kt) {
    v16h av;
#pragma unroll
    for (int h = 0; h < 16; ++h) {
      int kg = kt * 32 + akidx(h, hf);
      int m  = mt * 16 + l15;
      av[h] = (kg < 80) ? (_Float16)w[m * 80 + kg] : (_Float16)0.f;
    }
    A[kt] = av;
  }

  v8f acc[3] = {};   // one accumulator per owned N-tile

  for (int r = 0; r < RPA; ++r) {
    const int n = a * RPA + r;
    __syncthreads();   // protect bm from previous iteration's consumers

    // Staging with tap replication: wave wid stages rows {2*wid, 2*wid+1};
    // lanes stride columns -> coalesced global loads, one f32->f16 convert,
    // 5 LDS scatter stores (different im2col rows, 2-way bank conflicts max).
#pragma unroll
    for (int ii = 0; ii < 2; ++ii) {
      const int i = wid * 2 + ii;
      const float* xrow = &x[((size_t)n * CIN + i) * LL];
#pragma unroll
      for (int p = 0; p < 160; p += 32) {
        int c = p + lane;
        if (c < LL) {
          _Float16 v = (_Float16)xrow[c];
#pragma unroll
          for (int t = 0; t < 5; ++t) {
            int l = c + 2 - t;          // output col consuming x[c] at tap t
            if (l >= 0) bm[l][i * 5 + t] = v;   // l <= 151 < 160 always
          }
        }
      }
    }
    if (r + 1 < RPA)
      __builtin_prefetch(&x[((size_t)(n + 1) * CIN) * LL], 0, 0);
    __syncthreads();

#pragma unroll
    for (int qi = 0; qi < 3; ++qi) {
      int nt = bnt + 4 * qi;
      if (nt < 10) {                       // wave-uniform predicate
        v8f c = {};
#pragma unroll
        for (int kt = 0; kt < 3; ++kt) {
          // B element h: K = kt*32 + 16*hf + h, col = nt*16+l15 -> contiguous
          v16h b = ld16h(&bm[nt * 16 + l15][kt * 32 + 16 * hf]);
          c = wmma_f16(A[kt], b, c);
        }
#pragma unroll
        for (int j = 0; j < 8; ++j) acc[qi][j] += fmaxf(c[j], 0.f);  // per-read relu
      }
    }
  }

#pragma unroll
  for (int qi = 0; qi < 3; ++qi) {
    int nt = bnt + 4 * qi;
    int l  = nt * 16 + l15;
    if (nt < 10 && l < LL) {
#pragma unroll
      for (int j = 0; j < 8; ++j) {
        int m = mt * 16 + j + 8 * hf;
        red[((size_t)a * CC + m) * LL + l] = acc[qi][j];
      }
    }
  }
}

// ---------------------------------------------------------------------------
// Kernel 2: red2[n,:,l] = relu(comb_w[32,64] @ concat(red0,red1)[n,:,l])
// ---------------------------------------------------------------------------
__global__ __launch_bounds__(256)
void comb_kernel(const float* __restrict__ red0, const float* __restrict__ red1,
                 const float* __restrict__ cw, float* __restrict__ red2)
{
  const int n = blockIdx.x;
  const int tid = threadIdx.x, wid = tid >> 5, lane = tid & 31;
  const int hf = lane >> 4, l15 = lane & 15;

  v16h A[2][2];
#pragma unroll
  for (int mt = 0; mt < 2; ++mt)
#pragma unroll
    for (int kt = 0; kt < 2; ++kt) {
      v16h av;
#pragma unroll
      for (int h = 0; h < 16; ++h) {
        int kg = kt * 32 + akidx(h, hf);
        int m  = mt * 16 + l15;
        av[h] = (_Float16)cw[m * 64 + kg];
      }
      A[mt][kt] = av;
    }

  for (int nt = wid; nt < 10; nt += 8) {
    int l = nt * 16 + l15;
    v16h b0, b1;
#pragma unroll
    for (int h = 0; h < 16; ++h) {
      int K = h + 16 * hf;
      float v0 = 0.f, v1 = 0.f;
      if (l < LL) {
        v0 = red0[((size_t)n * CC + K) * LL + l];
        v1 = red1[((size_t)n * CC + K) * LL + l];
      }
      b0[h] = (_Float16)v0;
      b1[h] = (_Float16)v1;
    }
#pragma unroll
    for (int mt = 0; mt < 2; ++mt) {
      v8f c = {};
      c = wmma_f16(A[mt][0], b0, c);
      c = wmma_f16(A[mt][1], b1, c);
      if (l < LL) {
#pragma unroll
        for (int j = 0; j < 8; ++j) {
          int m = mt * 16 + j + 8 * hf;
          red2[((size_t)n * CC + m) * LL + l] = fmaxf(c[j], 0.f);
        }
      }
    }
  }
}

// ---------------------------------------------------------------------------
// Kernel 3: compressor = relu(conv1d(in, cw, k=3, stride=2, dil=2, pad=2))
// GEMM: M=32, K=96 (32ch x 3 taps), N=75->80. pairSum fuses reduce_slots(2).
// 5 waves per block, one N-tile each.
// ---------------------------------------------------------------------------
__global__ __launch_bounds__(160)
void compressor_kernel(const float* __restrict__ in, const float* __restrict__ cw,
                       float* __restrict__ out, int pairSum)
{
  const int n = blockIdx.x;
  const int tid = threadIdx.x, wid = tid >> 5, lane = tid & 31;
  const int hf = lane >> 4, l15 = lane & 15;
  const int nt = wid;   // 0..4

  v16h A[2][3];
#pragma unroll
  for (int mt = 0; mt < 2; ++mt)
#pragma unroll
    for (int kt = 0; kt < 3; ++kt) {
      v16h av;
#pragma unroll
      for (int h = 0; h < 16; ++h) {
        int kg = kt * 32 + akidx(h, hf);   // always < 96
        int m  = mt * 16 + l15;
        av[h] = (_Float16)cw[m * 96 + kg];
      }
      A[mt][kt] = av;
    }

  const float* inA = in + (size_t)(pairSum ? 2 * n : n) * CC * LL;
  const float* inB = inA + (size_t)CC * LL;   // used only when pairSum

  v16h b[3];
#pragma unroll
  for (int kt = 0; kt < 3; ++kt) {
#pragma unroll
    for (int h = 0; h < 16; ++h) {
      int kg = kt * 32 + h + 16 * hf;
      int i = kg / 3, t = kg % 3;
      int lp = nt * 16 + l15;
      int p  = 2 * lp + 2 * t - 2;
      float v = 0.f;
      if (lp < LCP && p >= 0 && p < LL) {
        v = inA[(size_t)i * LL + p];
        if (pairSum) v += inB[(size_t)i * LL + p];
      }
      b[kt][h] = (_Float16)v;
    }
  }

  const int lp = nt * 16 + l15;
#pragma unroll
  for (int mt = 0; mt < 2; ++mt) {
    v8f c = {};
#pragma unroll
    for (int kt = 0; kt < 3; ++kt) c = wmma_f16(A[mt][kt], b[kt], c);
    if (lp < LCP) {
#pragma unroll
      for (int j = 0; j < 8; ++j) {
        int m = mt * 16 + j + 8 * hf;
        out[((size_t)n * CC + m) * LCP + lp] = fmaxf(c[j], 0.f);
      }
    }
  }
}

// ---------------------------------------------------------------------------
// Kernel 3b: cfs1[s] = cfa[2s] + cfa[2s+1]   (reduce_slots over compressed)
// grid: (ceil(2400/256), NSITES) -> no div/mod in the kernel
// ---------------------------------------------------------------------------
__global__ void pairadd_kernel(const float* __restrict__ cfa, float* __restrict__ cfs1)
{
  const int e = blockIdx.x * blockDim.x + threadIdx.x;
  const int s = blockIdx.y;
  const size_t per = (size_t)CC * LCP;   // 2400
  if (e < (int)per)
    cfs1[(size_t)s * per + e] = cfa[(2 * s) * per + e] + cfa[(2 * s + 1) * per + e];
}

// ---------------------------------------------------------------------------
// Kernel 4: cross-attention + head, one wave per allele.
// q=Wq*cfa, kv=[exp0|exp1], k=Wk*kv, v=Wv*kv, P=softmax(q^T k/sqrt(C)),
// ctx=P v^T, e = mean_l(cfa+ctx) @ wo^T. All GEMMs 16x16x32 WMMA.
// qst/kst stored transposed (pos-major, 40-half stride) so every fragment
// load is 16B-aligned ds_load_b128 and every q/k store is ds_store_b128.
// ---------------------------------------------------------------------------
__global__ __launch_bounds__(32)
void xattn_kernel(const float* __restrict__ cfa, const float* __restrict__ cfs0,
                  const float* __restrict__ cfs1,
                  const float* __restrict__ wq, const float* __restrict__ wk,
                  const float* __restrict__ wvv, const float* __restrict__ wo,
                  float* __restrict__ eout)
{
  const int n    = blockIdx.x;
  const int lane = threadIdx.x & 31;
  const int hf   = lane >> 4, l15 = lane & 15;
  const int site = n >> 1;

  __shared__ __align__(16) _Float16 qst[80][40];    // q^T: [l][channel]
  __shared__ __align__(16) _Float16 kst[160][40];   // k^T: [m][channel]
  __shared__ __align__(16) _Float16 vs[CC][160];    // v:   [channel][m]
  __shared__ __align__(16) _Float16 ps[16][160];    // P tile: [row][m]
  __shared__ float rbuf[CC];

  // projection weight A-fragments (K = 32, single K-tile)
  v16h Aq[2], Ak[2], Av[2];
#pragma unroll
  for (int mt = 0; mt < 2; ++mt) {
    v16h aq, ak, av;
#pragma unroll
    for (int h = 0; h < 16; ++h) {
      int kg = akidx(h, hf);
      int m  = mt * 16 + l15;
      aq[h] = (_Float16)wq [m * CC + kg];
      ak[h] = (_Float16)wk [m * CC + kg];
      av[h] = (_Float16)wvv[m * CC + kg];
    }
    Aq[mt] = aq; Ak[mt] = ak; Av[mt] = av;
  }

  // q = Wq * cfa[n] -> qst[l][c]  (packed b128 stores: channels mt*16+8hf+j)
  for (int nt = 0; nt < 5; ++nt) {
    v16h b;
#pragma unroll
    for (int h = 0; h < 16; ++h) {
      int K = h + 16 * hf;
      int l = nt * 16 + l15;
      float v = (l < LCP) ? cfa[((size_t)n * CC + K) * LCP + l] : 0.f;
      b[h] = (_Float16)v;
    }
#pragma unroll
    for (int mt = 0; mt < 2; ++mt) {
      v8f z = {};
      v8f c = wmma_f16(Aq[mt], b, z);
      *(v8h*)&qst[nt * 16 + l15][mt * 16 + 8 * hf] = cvt8h(c);
    }
  }

  // k,v over kv = [cfs0[site] | cfs1[site]]; k -> kst[m][c], v -> vs[c][m]
  for (int nt = 0; nt < 10; ++nt) {
    v16h b;
#pragma unroll
    for (int h = 0; h < 16; ++h) {
      int K = h + 16 * hf;
      int m = nt * 16 + l15;
      float v = 0.f;
      if (m < LCP)            v = cfs0[((size_t)site * CC + K) * LCP + m];
      else if (m < 2 * LCP)   v = cfs1[((size_t)site * CC + K) * LCP + (m - LCP)];
      b[h] = (_Float16)v;
    }
#pragma unroll
    for (int mt = 0; mt < 2; ++mt) {
      v8f z = {};
      v8f ck = wmma_f16(Ak[mt], b, z);
      v8f cv = wmma_f16(Av[mt], b, z);
      *(v8h*)&kst[nt * 16 + l15][mt * 16 + 8 * hf] = cvt8h(ck);
#pragma unroll
      for (int j = 0; j < 8; ++j)
        vs[mt * 16 + j + 8 * hf][nt * 16 + l15] = (_Float16)cv[j];
    }
  }
  __syncthreads();

  float ctxsum0 = 0.f, ctxsum1 = 0.f;
  const float scale = 0.17677669529663687f;   // 1/sqrt(32)

  for (int lt = 0; lt < 5; ++lt) {
    // A = q^T tile: rows l = lt*16+l15; one contiguous 16-half channel run
    v16h aq = ld16h(&qst[lt * 16 + l15][16 * hf]);

    v8f sacc[10];
#pragma unroll
    for (int nt = 0; nt < 10; ++nt) {
      // B element h: channel = h+16hf at position m -> contiguous in kst
      v16h b = ld16h(&kst[nt * 16 + l15][16 * hf]);
      v8f z = {};
      sacc[nt] = wmma_f16(aq, b, z);
    }

    // per-row softmax (row spread over 16 lanes of one half, element j)
#pragma unroll
    for (int j = 0; j < 8; ++j) {
      float mx = -1e30f;
#pragma unroll
      for (int nt = 0; nt < 10; ++nt) {
        int col = nt * 16 + l15;
        if (col < 2 * LCP) mx = fmaxf(mx, sacc[nt][j] * scale);
      }
#pragma unroll
      for (int m = 8; m >= 1; m >>= 1) mx = fmaxf(mx, __shfl_xor(mx, m, 16));
      float ev[10], sm = 0.f;
#pragma unroll
      for (int nt = 0; nt < 10; ++nt) {
        int col = nt * 16 + l15;
        float e = (col < 2 * LCP) ? __expf(sacc[nt][j] * scale - mx) : 0.f;
        ev[nt] = e; sm += e;
      }
#pragma unroll
      for (int m = 8; m >= 1; m >>= 1) sm += __shfl_xor(sm, m, 16);
      float inv = 1.f / sm;
#pragma unroll
      for (int nt = 0; nt < 10; ++nt)
        ps[j + 8 * hf][nt * 16 + l15] = (_Float16)(ev[nt] * inv);
    }
    __syncthreads();

    // ctx tile (16 x 32) = P (16x160) @ v^T (160x32); all b128 fragment loads
#pragma unroll
    for (int ntc = 0; ntc < 2; ++ntc) {
      v8f cacc = {};
#pragma unroll
      for (int kt = 0; kt < 5; ++kt) {
        v16h ap = ld16h(&ps[l15][kt * 32 + 16 * hf]);
        v16h bv = ld16h(&vs[ntc * 16 + l15][kt * 32 + 16 * hf]);
        cacc = wmma_f16(ap, bv, cacc);
      }
#pragma unroll
      for (int j = 0; j < 8; ++j) {
        int l = lt * 16 + j + 8 * hf;
        if (l < LCP) { if (ntc == 0) ctxsum0 += cacc[j]; else ctxsum1 += cacc[j]; }
      }
    }
    __syncthreads();
  }

  // fold the two half-wave row groups, gather per-channel ctx sums
  ctxsum0 += __shfl_xor(ctxsum0, 16, 32);
  ctxsum1 += __shfl_xor(ctxsum1, 16, 32);
  if (hf == 0) { rbuf[l15] = ctxsum0; rbuf[16 + l15] = ctxsum1; }
  __syncthreads();

  // mean over l of (cfa + ctx), then head projection
  float cs = 0.f;
  for (int l = 0; l < LCP; ++l) cs += cfa[((size_t)n * CC + lane) * LCP + l];
  float meanv = (cs + rbuf[lane]) * (1.0f / LCP);
  __syncthreads();
  rbuf[lane] = meanv;
  __syncthreads();
  if (lane < 2) {
    float o = 0.f;
    for (int c = 0; c < CC; ++c) o += rbuf[c] * wo[lane * CC + c];
    eout[n * 2 + lane] = o;
  }
}

// ---------------------------------------------------------------------------
// Kernel 5: meta = softmax(mean_l([s0|s1]) @ meta_w^T + meta_b)
// ---------------------------------------------------------------------------
__global__ __launch_bounds__(64)
void meta_kernel(const float* __restrict__ s0, const float* __restrict__ s1,
                 const float* __restrict__ mw, const float* __restrict__ mb,
                 float* __restrict__ meta)
{
  const int s = blockIdx.x, c = threadIdx.x;
  __shared__ float feat[2 * CC];
  __shared__ float lg[3];
  const float* src = (c < CC) ? &s0[((size_t)s * CC + c) * LCP]
                              : &s1[((size_t)s * CC + (c - CC)) * LCP];
  float sum = 0.f;
  for (int l = 0; l < LCP; ++l) sum += src[l];
  feat[c] = sum * (1.0f / LCP);
  __syncthreads();
  if (c < 3) {
    float o = mb[c];
    for (int k = 0; k < 2 * CC; ++k) o += feat[k] * mw[c * 2 * CC + k];
    lg[c] = o;
  }
  __syncthreads();
  if (c == 0) {
    float mx = fmaxf(lg[0], fmaxf(lg[1], lg[2]));
    float e0 = __expf(lg[0] - mx), e1 = __expf(lg[1] - mx), e2 = __expf(lg[2] - mx);
    float inv = 1.f / (e0 + e1 + e2);
    meta[s * 3 + 0] = e0 * inv;
    meta[s * 3 + 1] = e1 * inv;
    meta[s * 3 + 2] = e2 * inv;
  }
}

// ---------------------------------------------------------------------------
extern "C" void kernel_launch(void* const* d_in, const int* in_sizes, int n_in,
                              void* d_out, int out_size, void* d_ws, size_t ws_size,
                              hipStream_t stream)
{
  const float* t0      = (const float*)d_in[0];
  const float* t1      = (const float*)d_in[1];
  const float* conv0_w = (const float*)d_in[2];
  const float* conv1_w = (const float*)d_in[3];
  const float* comp_w[3] = { (const float*)d_in[4], (const float*)d_in[5], (const float*)d_in[6] };
  const float* xq[3] = { (const float*)d_in[7],  (const float*)d_in[11], (const float*)d_in[15] };
  const float* xk[3] = { (const float*)d_in[8],  (const float*)d_in[12], (const float*)d_in[16] };
  const float* xv[3] = { (const float*)d_in[9],  (const float*)d_in[13], (const float*)d_in[17] };
  const float* xo[3] = { (const float*)d_in[10], (const float*)d_in[14], (const float*)d_in[18] };
  const float* comb_w = (const float*)d_in[19];
  const float* meta_w = (const float*)d_in[20];
  const float* meta_b = (const float*)d_in[21];
  float* out = (float*)d_out;

  // workspace layout (floats): 3x red [2048,32,150], cfa [2048,32,75],
  // cfs0/cfs1 [1024,32,75]  -> ~157 MB total
  float* ws = (float*)d_ws;
  const size_t redsz = (size_t)NALLELES * CC * LL;
  float* red[3];
  red[0] = ws;
  red[1] = red[0] + redsz;
  red[2] = red[1] + redsz;
  float* cfa  = red[2] + redsz;
  float* cfs0 = cfa  + (size_t)NALLELES * CC * LCP;
  float* cfs1 = cfs0 + (size_t)NSITES  * CC * LCP;

  conv_reduce_kernel<<<NALLELES, 256, 0, stream>>>(t0, conv0_w, red[0]);
  conv_reduce_kernel<<<NALLELES, 256, 0, stream>>>(t1, conv1_w, red[1]);
  comb_kernel<<<NALLELES, 256, 0, stream>>>(red[0], red[1], comb_w, red[2]);

  for (int s = 0; s < 3; ++s) {
    compressor_kernel<<<NALLELES, 160, 0, stream>>>(red[s], comp_w[s], cfa, 0);
    compressor_kernel<<<NSITES,   160, 0, stream>>>(red[s], comp_w[s], cfs0, 1);
    dim3 pg((CC * LCP + 255) / 256, NSITES);
    pairadd_kernel<<<pg, 256, 0, stream>>>(cfa, cfs1);
    xattn_kernel<<<NALLELES, 32, 0, stream>>>(cfa, cfs0, cfs1,
                                              xq[s], xk[s], xv[s], xo[s],
                                              out + (size_t)s * NALLELES * 2);
  }
  // stream 2's cfs0/cfs1 are still live here -> meta uses them
  meta_kernel<<<NSITES, 64, 0, stream>>>(cfs0, cfs1, meta_w, meta_b,
                                         out + 3 * (size_t)NALLELES * 2);
}
